// MultiHeadAttention_8478265442699
// MI455X (gfx1250) — matrix operand order repaired and checked by reading, not compiled, not run on hardware
//
#include <hip/hip_runtime.h>

// ---------------------------------------------------------------------------
// Problem constants (from the reference): B=2, T=2048, C=1024, H=16, D=64
// ---------------------------------------------------------------------------
#define BB 2
#define TT 2048
#define CC 1024
#define HH 16
#define DD 64
#define MM (BB * TT)          // 4096 tokens
#define NEGV (-1e9f)

typedef __attribute__((ext_vector_type(16))) __bf16 v16bf;
typedef __attribute__((ext_vector_type(8)))  float  v8f;
typedef unsigned int tdm_u32x4 __attribute__((ext_vector_type(4)));
typedef int          tdm_i32x8 __attribute__((ext_vector_type(8)));
typedef int          tdm_i32x4 __attribute__((ext_vector_type(4)));

union FragAB { v16bf v; unsigned short u[16]; };
union FragC  { v8f   v; float f[8]; };

__device__ __forceinline__ unsigned short f32_to_bf16(float f) {
    // gfx1250 has native bf16 converts (v_cvt_pk_bf16_f32); let clang use them
    __bf16 h = (__bf16)f;
    return __builtin_bit_cast(unsigned short, h);
}

// ---------------------------------------------------------------------------
// Tensor Data Mover: DMA a 2-D bf16 tile (tile_w x tile_h elements, row
// stride row_stride elements) from global memory into LDS at lds_off.
// Descriptor layout per CDNA5 ISA ch.8 (D# group0/group1); groups 2/3 zero
// (2-D tensor, tile_dim2=0 => unused). Issued once per wave; TENSORcnt.
// ---------------------------------------------------------------------------
__device__ __forceinline__ void tdm_load_2d_bf16(unsigned lds_off,
                                                 const unsigned short* gptr,
                                                 unsigned tile_w,
                                                 unsigned tile_h,
                                                 unsigned row_stride) {
    unsigned long long ga = (unsigned long long)(size_t)gptr;
    tdm_u32x4 g0;
    g0.x = 1u;                                    // count=1, no gather
    g0.y = lds_off;                               // lds_addr
    g0.z = (unsigned)(ga & 0xFFFFFFFFu);          // global_addr[31:0]
    g0.w = (unsigned)((ga >> 32) & 0x1FFFFFFu)    // global_addr[56:32]
         | (2u << 30);                            // type=2 ("image")
    const unsigned td0 = 1u << 20;                // tensor_dim0 (no OOB clip)
    const unsigned td1 = 1u << 20;                // tensor_dim1
    const unsigned long long s0 = row_stride;     // tensor_dim0_stride
    tdm_i32x8 g1;
    g1[0] = (int)(1u << 16);                      // data_size=1 (2 bytes)
    g1[1] = (int)((td0 & 0xFFFFu) << 16);         // td0[15:0]
    g1[2] = (int)((td0 >> 16) | ((td1 & 0xFFFFu) << 16));
    g1[3] = (int)((td1 >> 16) | (tile_w << 16));  // tile_dim0
    g1[4] = (int)(tile_h & 0xFFFFu);              // tile_dim1 (tile_dim2=0)
    g1[5] = (int)(unsigned)(s0 & 0xFFFFFFFFu);    // stride0[31:0]
    g1[6] = (int)(unsigned)((s0 >> 32) & 0xFFFFu);// stride0[47:32]
    g1[7] = 0;                                    // stride1 (unused, 2-D)
    tdm_i32x4 z4 = {0, 0, 0, 0};
#if __clang_major__ >= 23
    tdm_i32x8 z8 = {0, 0, 0, 0, 0, 0, 0, 0};
    __builtin_amdgcn_tensor_load_to_lds(g0, g1, z4, z4, z8, 0);
#else
    __builtin_amdgcn_tensor_load_to_lds(g0, g1, z4, z4, 0);
#endif
}

__device__ __forceinline__ unsigned lds_offset_of(const void* p) {
    // Generic->LDS: low 32 bits of the generic address are the LDS offset.
    return (unsigned)(unsigned long long)(size_t)p;
}

// ---------------------------------------------------------------------------
// f32 -> bf16 conversion
// ---------------------------------------------------------------------------
__global__ void cvt_f32_bf16_kernel(const float* __restrict__ in,
                                    unsigned short* __restrict__ out, int n) {
    int i = blockIdx.x * blockDim.x + threadIdx.x;
    if (i < n) out[i] = f32_to_bf16(in[i]);
}

// ---------------------------------------------------------------------------
// GEMM: Out = A[M,K] @ B[N,K]^T  (bf16 inputs, f32 accumulate)
// block = 256 threads (8 waves); each wave computes a 32x64 strip.
// B tile (64 x 32k) is DMA'd into LDS by the TDM, double-buffered so the
// transfer for step s+1 overlaps the 8 WMMAs of step s.
// TRANS_OUT=false: Out is [M,N].  TRANS_OUT=true: Out is [N,M] (bf16 only) —
// used for V so the attention B-operand (V^T) becomes contiguous.
// DO_SCALE: multiply result by `scale` (used to fold 1/sqrt(D) into Q).
// grid = (N/64, M/256)
// ---------------------------------------------------------------------------
template <bool OUT_BF16, bool TRANS_OUT, bool DO_SCALE>
__global__ __launch_bounds__(256) void
gemm_bf16_abt_kernel(const unsigned short* __restrict__ A,
                     const unsigned short* __restrict__ Bm,
                     void* __restrict__ Out, int M, int N, int K, float scale) {
    const int lane = threadIdx.x & 31;
    const int wave = threadIdx.x >> 5;
    const int lo = lane & 15;
    const int hi = lane >> 4;
    const int m0 = blockIdx.y * 256 + wave * 32;
    const int n0 = blockIdx.x * 64;

    __shared__ __align__(16) unsigned short Bt[2][64 * 32];

    FragC acc[2][4];
#pragma unroll
    for (int r = 0; r < 2; ++r)
#pragma unroll
        for (int nt = 0; nt < 4; ++nt)
#pragma unroll
            for (int j = 0; j < 8; ++j) acc[r][nt].f[j] = 0.0f;

    const unsigned short* arow0 = A + (size_t)(m0 + lo) * K;
    const unsigned short* arow1 = A + (size_t)(m0 + 16 + lo) * K;
    const unsigned short* bbase = Bm + (size_t)n0 * K;
    const int nsteps = K / 32;

    if (wave == 0)
        tdm_load_2d_bf16(lds_offset_of(&Bt[0][0]), bbase, 32, 64, K);

    for (int s = 0; s < nsteps; ++s) {
        if (wave == 0) {
            if (s + 1 < nsteps) {
                tdm_load_2d_bf16(lds_offset_of(&Bt[(s + 1) & 1][0]),
                                 bbase + (s + 1) * 32, 32, 64, K);
                __builtin_amdgcn_s_wait_tensorcnt(1);  // oldest tile done
            } else {
                __builtin_amdgcn_s_wait_tensorcnt(0);
            }
        }
        __syncthreads();

        const int k0 = s * 32;
        // A fragments: lane = row; elems 0..7 -> k=base+0..7, 8..15 -> +16..23
        FragAB a[2];
        {
            const unsigned short* pa0 = arow0 + k0 + hi * 8;
            const unsigned short* pa1 = arow1 + k0 + hi * 8;
            if (k0 + 32 < K) {
                __builtin_prefetch(arow0 + k0 + 32, 0, 0);
                __builtin_prefetch(arow1 + k0 + 32, 0, 0);
            }
#pragma unroll
            for (int e = 0; e < 8; ++e) {
                a[0].u[e]     = pa0[e];
                a[0].u[8 + e] = pa0[16 + e];
                a[1].u[e]     = pa1[e];
                a[1].u[8 + e] = pa1[16 + e];
            }
        }
        const unsigned short* bt = &Bt[s & 1][0];
#pragma unroll
        for (int nt = 0; nt < 4; ++nt) {
            // B fragment: lane = column; elems 0..7 -> k=hi*8+0..7, 8..15 -> +16..23
            FragAB b;
            const unsigned short* pb = bt + (nt * 16 + lo) * 32 + hi * 8;
#pragma unroll
            for (int e = 0; e < 8; ++e) { b.u[e] = pb[e]; b.u[8 + e] = pb[16 + e]; }
#pragma unroll
            for (int r = 0; r < 2; ++r)
                acc[r][nt].v = __builtin_amdgcn_wmma_f32_16x16x32_bf16(
                    false, a[r].v, false, b.v, (short)0, acc[r][nt].v,
                    false, false);
        }
        __syncthreads();   // all waves done with Bt[s&1] before TDM reuses it
    }

    if (DO_SCALE) {
#pragma unroll
        for (int r = 0; r < 2; ++r)
#pragma unroll
            for (int nt = 0; nt < 4; ++nt)
#pragma unroll
                for (int j = 0; j < 8; ++j) acc[r][nt].f[j] *= scale;
    }

    // C layout: row = m0 + r*16 + j + 8*hi, col = n0 + nt*16 + lo
    if (TRANS_OUT) {
        // Out[N][M] bf16; per lane each (r,nt) is an 8-element contiguous run
        unsigned short* O = (unsigned short*)Out;
#pragma unroll
        for (int r = 0; r < 2; ++r)
#pragma unroll
            for (int nt = 0; nt < 4; ++nt)
#pragma unroll
                for (int j = 0; j < 8; ++j)
                    O[(size_t)(n0 + nt * 16 + lo) * M + m0 + r * 16 + 8 * hi + j] =
                        f32_to_bf16(acc[r][nt].f[j]);
    } else if (OUT_BF16) {
        unsigned short* O = (unsigned short*)Out;
#pragma unroll
        for (int r = 0; r < 2; ++r)
#pragma unroll
            for (int nt = 0; nt < 4; ++nt)
#pragma unroll
                for (int j = 0; j < 8; ++j)
                    O[(size_t)(m0 + r * 16 + j + 8 * hi) * N + n0 + nt * 16 + lo] =
                        f32_to_bf16(acc[r][nt].f[j]);
    } else {
        float* O = (float*)Out;
#pragma unroll
        for (int r = 0; r < 2; ++r)
#pragma unroll
            for (int nt = 0; nt < 4; ++nt)
#pragma unroll
                for (int j = 0; j < 8; ++j)
                    O[(size_t)(m0 + r * 16 + j + 8 * hi) * N + n0 + nt * 16 + lo] =
                        acc[r][nt].f[j];
    }
}

// ---------------------------------------------------------------------------
// Flash-attention: one block per (b, h, 64-query tile); 128 threads = 4 waves,
// each wave owns 16 query rows. K (row-major [B*T,C]) and V^T ([C,B*T]) tiles
// are DMA'd into double-buffered LDS by the TDM; online softmax. Q is
// pre-scaled by 1/sqrt(D). Causal compare only on the diagonal KV tile; all
// earlier tiles are fully below the diagonal for this query block.
// ctx out: bf16 [B*T, C].
// ---------------------------------------------------------------------------
__global__ __launch_bounds__(128) void
attn_kernel(const unsigned short* __restrict__ Q,
            const unsigned short* __restrict__ Kc,
            const unsigned short* __restrict__ Vtg,   // V^T [C][B*T]
            const unsigned char* __restrict__ pad,
            unsigned short* __restrict__ ctx) {
    const int tiles = TT / 64;
    const int bh = blockIdx.x / tiles;
    const int qt = blockIdx.x % tiles;
    const int b = bh / HH;
    const int h = bh % HH;

    const int lane = threadIdx.x & 31;
    const int wave = threadIdx.x >> 5;
    const int lo = lane & 15;
    const int hi = lane >> 4;
    const int q0 = qt * 64 + wave * 16;                // wave's first query row
    const size_t base = ((size_t)b * TT) * CC + (size_t)h * DD;

    __shared__ __align__(16) unsigned short Kt[2][64 * 64];  // [key][d]
    __shared__ __align__(16) unsigned short Vt[2][64 * 64];  // [d][key]
    __shared__ __align__(16) unsigned short Ps[4][16 * 64];

    // Load Q fragments for this wave's 16 rows (kept in registers all loop).
    FragAB qf[2];
    {
        const unsigned short* qrow = Q + base + (size_t)(q0 + lo) * CC;
#pragma unroll
        for (int kk = 0; kk < 2; ++kk) {
            const unsigned short* p = qrow + kk * 32 + hi * 8;
#pragma unroll
            for (int e = 0; e < 8; ++e) {
                qf[kk].u[e]     = p[e];
                qf[kk].u[8 + e] = p[16 + e];
            }
        }
    }

    float mrow[8], lrow[8];
    FragC oacc[4];
#pragma unroll
    for (int j = 0; j < 8; ++j) { mrow[j] = -3e38f; lrow[j] = 0.0f; }
#pragma unroll
    for (int nt = 0; nt < 4; ++nt)
#pragma unroll
        for (int j = 0; j < 8; ++j) oacc[nt].f[j] = 0.0f;

    const int nkv = qt + 1;                     // causal: kv tiles 0..qt
    const unsigned short* kg = Kc + base;
    const unsigned short* vg = Vtg + (size_t)(h * DD) * MM + (size_t)b * TT;

    if (wave == 0) {
        tdm_load_2d_bf16(lds_offset_of(&Kt[0][0]), kg, 64, 64, CC);
        tdm_load_2d_bf16(lds_offset_of(&Vt[0][0]), vg, 64, 64, MM);
    }

    for (int it = 0; it < nkv; ++it) {
        const int kv0 = it * 64;
        if (wave == 0) {
            if (it + 1 < nkv) {
                tdm_load_2d_bf16(lds_offset_of(&Kt[(it + 1) & 1][0]),
                                 kg + (size_t)(kv0 + 64) * CC, 64, 64, CC);
                tdm_load_2d_bf16(lds_offset_of(&Vt[(it + 1) & 1][0]),
                                 vg + (kv0 + 64), 64, 64, MM);
                __builtin_amdgcn_s_wait_tensorcnt(2);  // current pair done
            } else {
                __builtin_amdgcn_s_wait_tensorcnt(0);
            }
        }
        __syncthreads();

        const unsigned short* kt = &Kt[it & 1][0];
        const unsigned short* vt = &Vt[it & 1][0];

        // S = Q * K^T  (16x64 strip per wave; 8 WMMAs)
        FragC s[4];
#pragma unroll
        for (int nt = 0; nt < 4; ++nt)
#pragma unroll
            for (int j = 0; j < 8; ++j) s[nt].f[j] = 0.0f;
#pragma unroll
        for (int nt = 0; nt < 4; ++nt) {
#pragma unroll
            for (int kk = 0; kk < 2; ++kk) {
                FragAB bfr;
                const unsigned short* p =
                    kt + (nt * 16 + lo) * 64 + kk * 32 + hi * 8;
#pragma unroll
                for (int e = 0; e < 8; ++e) { bfr.u[e] = p[e]; bfr.u[8 + e] = p[16 + e]; }
                s[nt].v = __builtin_amdgcn_wmma_f32_16x16x32_bf16(
                    false, qf[kk].v, false, bfr.v, (short)0, s[nt].v,
                    false, false);
            }
        }

        // masking (Q already carries the 1/sqrt(D) scale)
        if (it == qt) {
            // diagonal tile: causal compare + pad mask, per element
#pragma unroll
            for (int nt = 0; nt < 4; ++nt) {
                const int key = kv0 + nt * 16 + lo;
                const bool padded = (pad[(size_t)b * TT + key] == 0);
#pragma unroll
                for (int j = 0; j < 8; ++j) {
                    const int qa = q0 + j + 8 * hi;
                    if (key > qa || padded) s[nt].f[j] = NEGV;
                }
            }
        } else {
            // strictly-lower tile: every key precedes every query; pad only
#pragma unroll
            for (int nt = 0; nt < 4; ++nt) {
                const int key = kv0 + nt * 16 + lo;
                const bool padded = (pad[(size_t)b * TT + key] == 0);
#pragma unroll
                for (int j = 0; j < 8; ++j)
                    if (padded) s[nt].f[j] = NEGV;
            }
        }

        // online softmax per row (row j+8*hi replicated across 16 lanes)
#pragma unroll
        for (int j = 0; j < 8; ++j) {
            float r = s[0].f[j];
#pragma unroll
            for (int nt = 1; nt < 4; ++nt) r = fmaxf(r, s[nt].f[j]);
#pragma unroll
            for (int msk = 1; msk < 16; msk <<= 1)
                r = fmaxf(r, __shfl_xor(r, msk, 32));
            const float mn = fmaxf(mrow[j], r);
            const float al = __expf(mrow[j] - mn);
            float rs = 0.0f;
#pragma unroll
            for (int nt = 0; nt < 4; ++nt) {
                float pv = __expf(s[nt].f[j] - mn);
                s[nt].f[j] = pv;
                rs += pv;
            }
#pragma unroll
            for (int msk = 1; msk < 16; msk <<= 1)
                rs += __shfl_xor(rs, msk, 32);
            lrow[j] = lrow[j] * al + rs;
            mrow[j] = mn;
#pragma unroll
            for (int nt = 0; nt < 4; ++nt) oacc[nt].f[j] *= al;
        }

        // stage P (bf16) through this wave's private LDS slab to re-layout
        // from C-fragment order into A-fragment order
#pragma unroll
        for (int nt = 0; nt < 4; ++nt)
#pragma unroll
            for (int j = 0; j < 8; ++j)
                Ps[wave][(j + 8 * hi) * 64 + nt * 16 + lo] =
                    f32_to_bf16(s[nt].f[j]);

        FragAB pf[2];
#pragma unroll
        for (int kk = 0; kk < 2; ++kk) {
            const unsigned short* p = &Ps[wave][lo * 64 + kk * 32 + hi * 8];
#pragma unroll
            for (int e = 0; e < 8; ++e) {
                pf[kk].u[e]     = p[e];
                pf[kk].u[8 + e] = p[16 + e];
            }
        }

        // O += P * V (8 WMMAs); V^T tile makes B-fragments contiguous:
        // lane = column d = nt*16+lo, elements = keys hi*8+0..7 and +16..23
#pragma unroll
        for (int nt = 0; nt < 4; ++nt) {
#pragma unroll
            for (int kk = 0; kk < 2; ++kk) {
                FragAB vb;
                const unsigned short* p =
                    vt + (nt * 16 + lo) * 64 + kk * 32 + hi * 8;
#pragma unroll
                for (int e = 0; e < 8; ++e) { vb.u[e] = p[e]; vb.u[8 + e] = p[16 + e]; }
                oacc[nt].v = __builtin_amdgcn_wmma_f32_16x16x32_bf16(
                    false, pf[kk].v, false, vb.v, (short)0, oacc[nt].v,
                    false, false);
            }
        }
        __syncthreads();   // all waves done with Kt/Vt[it&1] before TDM reuse
    }

    // normalize and write ctx (bf16, layout [B*T, C] with head at col h*64)
#pragma unroll
    for (int nt = 0; nt < 4; ++nt)
#pragma unroll
        for (int j = 0; j < 8; ++j) {
            const float o = oacc[nt].f[j] / lrow[j];
            ctx[base + (size_t)(q0 + j + 8 * hi) * CC + nt * 16 + lo] =
                f32_to_bf16(o);
        }
}

// ---------------------------------------------------------------------------
// Launch
// ---------------------------------------------------------------------------
extern "C" void kernel_launch(void* const* d_in, const int* in_sizes, int n_in,
                              void* d_out, int out_size, void* d_ws,
                              size_t ws_size, hipStream_t stream) {
    const float* x = (const float*)d_in[0];
    const unsigned char* pad = (const unsigned char*)d_in[1];
    const float* Wq = (const float*)d_in[2];
    const float* Wk = (const float*)d_in[3];
    const float* Wv = (const float*)d_in[4];
    const float* Wo = (const float*)d_in[5];

    const int M = MM;                        // 4096
    const size_t MC = (size_t)M * CC;        // x / Q / K / V / ctx elems
    const size_t WC = (size_t)CC * CC;       // weight elems

    unsigned short* xb   = (unsigned short*)d_ws;
    unsigned short* wqb  = xb + MC;
    unsigned short* wkb  = wqb + WC;
    unsigned short* wvb  = wkb + WC;
    unsigned short* wob  = wvb + WC;
    unsigned short* Qb   = wob + WC;
    unsigned short* Kb   = Qb + MC;
    unsigned short* Vtb  = Kb + MC;          // V^T [C][M]
    unsigned short* ctxb = Vtb + MC;

    // 1) convert inputs to bf16
    cvt_f32_bf16_kernel<<<(int)((MC + 255) / 256), 256, 0, stream>>>(x, xb, (int)MC);
    cvt_f32_bf16_kernel<<<(int)((WC + 255) / 256), 256, 0, stream>>>(Wq, wqb, (int)WC);
    cvt_f32_bf16_kernel<<<(int)((WC + 255) / 256), 256, 0, stream>>>(Wk, wkb, (int)WC);
    cvt_f32_bf16_kernel<<<(int)((WC + 255) / 256), 256, 0, stream>>>(Wv, wvb, (int)WC);
    cvt_f32_bf16_kernel<<<(int)((WC + 255) / 256), 256, 0, stream>>>(Wo, wob, (int)WC);

    // 2) Q/K/V projections: y = x @ W^T (bf16 out; Q pre-scaled by 1/sqrt(D);
    //    V written transposed)
    dim3 ggrid(CC / 64, M / 256);
    gemm_bf16_abt_kernel<true, false, true ><<<ggrid, 256, 0, stream>>>(
        xb, wqb, Qb, M, CC, CC, 0.125f);
    gemm_bf16_abt_kernel<true, false, false><<<ggrid, 256, 0, stream>>>(
        xb, wkb, Kb, M, CC, CC, 1.0f);
    gemm_bf16_abt_kernel<true, true,  false><<<ggrid, 256, 0, stream>>>(
        xb, wvb, Vtb, M, CC, CC, 1.0f);

    // 3) attention
    attn_kernel<<<BB * HH * (TT / 64), 128, 0, stream>>>(Qb, Kb, Vtb, pad, ctxb);

    // 4) output projection: out = ctx @ Wo^T (f32 out)
    gemm_bf16_abt_kernel<false, false, false><<<ggrid, 256, 0, stream>>>(
        ctxb, wob, (float*)d_out, M, CC, CC, 1.0f);
}
